// CausalSelfAttention_40003325395456
// MI455X (gfx1250) — matrix operand order repaired and checked by reading, not compiled, not run on hardware
//
#include <hip/hip_runtime.h>
#include <hip/hip_bf16.h>

typedef __bf16 bf16_t;
typedef __attribute__((ext_vector_type(16))) __bf16 v16bf;
typedef __attribute__((ext_vector_type(8)))  float   v8f;
typedef __attribute__((ext_vector_type(4)))  unsigned int u32x4;
typedef __attribute__((ext_vector_type(4)))  int i32x4;

static constexpr int Bq = 4, Hh = 8, Ll = 4096, Dd = 64, Ee = 512;
static constexpr int Nrows = Bq * Ll;        // 16384
static constexpr int CH = 128, NC = Ll / CH; // 32 chunks
// LDS leading dims chosen for 16B alignment + conflict-free WMMA fragment reads
static constexpr int QLD = 72;   // row-major bf16 [128][72]; also state [64][72]
static constexpr int CHP = 136;  // transposed bf16 [64][136] / scores [128][136]
#define EPSF 1e-6f

#define DEVFN static __device__ __forceinline__

#if __has_builtin(__builtin_amdgcn_global_load_async_to_lds_b128)
#define HAVE_ASYNC 1
#else
#define HAVE_ASYNC 0
#endif

// async-to-LDS builtin wants (int4 AS1*, int4 AS3*, imm offset, imm cpol)
typedef __attribute__((address_space(1))) i32x4 glb_v4i;
typedef __attribute__((address_space(3))) i32x4 lds_v4i;
DEVFN glb_v4i* to_glb(const void* p) {
  return (glb_v4i*)(unsigned long long)p;
}
DEVFN lds_v4i* to_lds(void* p) {
  return (lds_v4i*)(unsigned)(unsigned long long)p;
}
DEVFN void wait_async_done() {
#if HAVE_ASYNC
#if __has_builtin(__builtin_amdgcn_s_wait_asynccnt)
  __builtin_amdgcn_s_wait_asynccnt(0);
#else
  asm volatile("s_wait_asynccnt 0" ::: "memory");
#endif
#endif
}

// ---------------- WMMA fragment helpers (CDNA5 wave32 layouts) ----------------
// A 16x32 bf16: lane (r=lane&15, h=lane>>4): f[0..7]=cols 8h.., f[8..15]=cols 16+8h..
DEVFN v16bf frag_a_rows(const bf16_t* p, int ld, int lane) {
  int r = lane & 15, h = (lane >> 4) & 1;
  const bf16_t* row = p + (size_t)r * ld + 8 * h;
  union { v16bf v; u32x4 q[2]; } u;
  u.q[0] = *(const u32x4*)(row);
  u.q[1] = *(const u32x4*)(row + 16);
  return u.v;
}
// B 32x16 bf16: lane (n=lane&15, h=lane>>4) holds column n, f[0..7]=K 8h.., f[8..15]=K 16+8h..
// Source where B[kk,nn] = p[nn*ld + kk]
DEVFN v16bf frag_b_rows(const bf16_t* p, int ld, int lane) {
  int n = lane & 15, h = (lane >> 4) & 1;
  const bf16_t* row = p + (size_t)n * ld + 8 * h;
  union { v16bf v; u32x4 q[2]; } u;
  u.q[0] = *(const u32x4*)(row);
  u.q[1] = *(const u32x4*)(row + 16);
  return u.v;
}
DEVFN v8f wmma_bf16(v16bf a, v16bf b, v8f c) {
  return __builtin_amdgcn_wmma_f32_16x16x32_bf16(false, a, false, b, (short)0, c,
                                                 false, false);
}

// ---------------- f32 -> bf16 conversion ----------------
__global__ void f32_to_bf16_kernel(const float* __restrict__ src,
                                   bf16_t* __restrict__ dst, int n) {
  int i = blockIdx.x * blockDim.x + threadIdx.x;
  if (i < n) dst[i] = (bf16_t)src[i];
}

// ---------------- GEMM: Y = X @ W^T + b  (X:[N,512] bf16, W:[512,512] bf16) ---
template <bool OUT_HEADS, int ACT>
__global__ __launch_bounds__(256) void gemm_bf16_kernel(
    const bf16_t* __restrict__ X, const bf16_t* __restrict__ W,
    const float* __restrict__ bias, float* __restrict__ Y) {
  int lane = threadIdx.x & 31, wid = threadIdx.x >> 5;
  int rw = wid & 3, cw = wid >> 2;
  int row0 = blockIdx.x * 128 + rw * 32;
  int col0 = blockIdx.y * 64 + cw * 32;
  v8f c[2][2] = {};
  const bf16_t* Xp = X + (size_t)row0 * Ee;
  const bf16_t* Wp = W + (size_t)col0 * Ee;
#pragma unroll 4
  for (int kk = 0; kk < Ee; kk += 32) {
    __builtin_prefetch(Xp + kk + 256, 0, 1);  // global_prefetch_b8
    v16bf a0 = frag_a_rows(Xp + kk, Ee, lane);
    v16bf a1 = frag_a_rows(Xp + 16 * Ee + kk, Ee, lane);
    v16bf b0 = frag_b_rows(Wp + kk, Ee, lane);
    v16bf b1 = frag_b_rows(Wp + 16 * Ee + kk, Ee, lane);
    c[0][0] = wmma_bf16(a0, b0, c[0][0]);
    c[0][1] = wmma_bf16(a0, b1, c[0][1]);
    c[1][0] = wmma_bf16(a1, b0, c[1][0]);
    c[1][1] = wmma_bf16(a1, b1, c[1][1]);
  }
  int coll = lane & 15, rhalf = (lane >> 4) & 1;
#pragma unroll
  for (int tr = 0; tr < 2; ++tr)
#pragma unroll
    for (int tc = 0; tc < 2; ++tc) {
      int gcol = col0 + tc * 16 + coll;
      float bv = bias[gcol];
      int rbase = row0 + tr * 16 + rhalf * 8;
#pragma unroll
      for (int rr = 0; rr < 8; ++rr) {
        float val = c[tr][tc][rr] + bv;
        if (ACT == 1) val = 1.0f / (1.0f + __expf(-val));
        int grow = rbase + rr;
        if (OUT_HEADS) {
          int bb = grow >> 12, l = grow & (Ll - 1);
          int hh = gcol >> 6, dd = gcol & 63;
          Y[(((size_t)bb * Hh + hh) * Ll + l) * Dd + dd] = val;
        } else {
          Y[(size_t)grow * Ee + gcol] = val;
        }
      }
    }
}

// ---------------- Flow-attention normalization scan ----------------
__global__ __launch_bounds__(32) void flow_scan_kernel(
    const float* __restrict__ qf, const float* __restrict__ kf,
    const float* __restrict__ vf, bf16_t* __restrict__ qm,
    bf16_t* __restrict__ kb, bf16_t* __restrict__ vm,
    float* __restrict__ sink) {
  int bh = blockIdx.x, lane = threadIdx.x;
  const size_t base = (size_t)bh * Ll * Dd;
  const float* q = qf + base;
  const float* k = kf + base;
  const float* v = vf + base;
  float kc0 = 0.f, kc1 = 0.f, qc0 = 0.f, qc1 = 0.f;
  float s10 = 0.f, s11 = 0.f, s20 = 0.f, s21 = 0.f, cssum = 0.f;
  for (int l = 0; l < Ll; ++l) {
    size_t off = (size_t)l * Dd + 2 * lane;
    float2 qv = *(const float2*)(q + off);
    float2 kv = *(const float2*)(k + off);
    kc0 += kv.x; kc1 += kv.y; qc0 += qv.x; qc1 += qv.y;
    float d1 = (qv.x + EPSF) * (kc0 + EPSF) + (qv.y + EPSF) * (kc1 + EPSF);
    float d2 = (kv.x + EPSF) * (qc0 + EPSF) + (kv.y + EPSF) * (qc1 + EPSF);
#pragma unroll
    for (int m = 16; m; m >>= 1) { d1 += __shfl_xor(d1, m, 32); d2 += __shfl_xor(d2, m, 32); }
    float normal = (float)(l + 1);
    float sink_in = normal / d1;   // sink_incoming * normal
    float src_out = normal / d2;   // source_outgoing * normal
    s10 += kv.x * src_out; s11 += kv.y * src_out;
    s20 += qv.x * sink_in; s21 += qv.y * sink_in;
    float d3 = (qv.x + EPSF) * (s10 + EPSF) + (qv.y + EPSF) * (s11 + EPSF);
    float d4 = (kv.x + EPSF) * (s20 + EPSF) + (kv.y + EPSF) * (s21 + EPSF);
#pragma unroll
    for (int m = 16; m; m >>= 1) { d3 += __shfl_xor(d3, m, 32); d4 += __shfl_xor(d4, m, 32); }
    float csink = d3 / normal;
    float csrc  = fminf(fmaxf(d4 / normal, -1.f), 1.f);
    float salloc = 1.f / (1.f + __expf(-csink));
    float cse = __expf(csrc);
    cssum += cse;
    float scomp = cse / cssum * normal;  // source_competition
    float qs = sink_in / normal;         // sink_incoming / normal
    qm[base + off]     = (bf16_t)(qv.x * qs);
    qm[base + off + 1] = (bf16_t)(qv.y * qs);
    kb[base + off]     = (bf16_t)kv.x;
    kb[base + off + 1] = (bf16_t)kv.y;
    float2 vv = *(const float2*)(v + off);
    vm[base + off]     = (bf16_t)(vv.x * scomp);
    vm[base + off + 1] = (bf16_t)(vv.y * scomp);
    if (lane == 0) sink[(size_t)bh * Ll + l] = salloc;
  }
}

// Stage a row-major [128][64] bf16 global chunk as transposed [64][CHP] in LDS.
DEVFN void stage_transposed(const bf16_t* __restrict__ g, bf16_t* lds, int tid) {
  for (int i = tid; i < CH * Dd / 8; i += 256) {
    int l = i >> 3, m0 = (i & 7) * 8;
    union { u32x4 q; bf16_t e[8]; } u;
    u.q = ((const u32x4*)g)[i];
#pragma unroll
    for (int e2 = 0; e2 < 8; ++e2) lds[(m0 + e2) * CHP + l] = u.e[e2];
  }
}

// ---------------- Per-chunk kvT = V^T K  ([m][d] f32 per (bh,chunk)) ---------
__global__ __launch_bounds__(256) void kv_chunk_kernel(
    const bf16_t* __restrict__ kb, const bf16_t* __restrict__ vm,
    float* __restrict__ kvws) {
  __shared__ bf16_t Kt[Dd * CHP];
  __shared__ bf16_t Vt[Dd * CHP];
  int tid = threadIdx.x, lane = tid & 31, wid = tid >> 5;
  int bh = blockIdx.x / NC, n = blockIdx.x % NC;
  const size_t src = ((size_t)bh * Ll + (size_t)n * CH) * Dd;
  stage_transposed(kb + src, Kt, tid);
  stage_transposed(vm + src, Vt, tid);
  __syncthreads();
  float* out = kvws + (size_t)blockIdx.x * Dd * Dd;
#pragma unroll
  for (int t = 0; t < 2; ++t) {
    int ti = wid * 2 + t, mT = ti >> 2, dT = ti & 3;
    v8f c = {};
#pragma unroll
    for (int l0 = 0; l0 < CH; l0 += 32) {
      v16bf a = frag_a_rows(Vt + mT * 16 * CHP + l0, CHP, lane);   // A[m,l]=V[l,m]
      v16bf b = frag_b_rows(Kt + dT * 16 * CHP + l0, CHP, lane);   // B[l,d]=K[l,d]
      c = wmma_bf16(a, b, c);
    }
    int coll = lane & 15, rhalf = (lane >> 4) & 1;
#pragma unroll
    for (int rr = 0; rr < 8; ++rr) {
      int m = mT * 16 + rhalf * 8 + rr;
      out[(size_t)m * Dd + dT * 16 + coll] = c[rr];  // kvT[m][d]
    }
  }
}

// ---------------- Per-chunk output: Q@state + tril(QK^T)@V, * sink_alloc -----
__global__ __launch_bounds__(256) void attn_chunk_kernel(
    const bf16_t* __restrict__ qm, const bf16_t* __restrict__ kb,
    const bf16_t* __restrict__ vm, const float* __restrict__ kvws,
    const float* __restrict__ sink, bf16_t* __restrict__ attnb) {
  extern __shared__ char smem[];
  bf16_t* Qs = (bf16_t*)smem;                 // [128][QLD]
  bf16_t* Ks = Qs + CH * QLD;                 // [128][QLD]
  bf16_t* Vt = Ks + CH * QLD;                 // [64][CHP]  (V transposed)
  bf16_t* Sc = Vt + Dd * CHP;                 // [128][CHP] masked scores
  bf16_t* statebf = Sc + CH * CHP;            // [64][QLD]  state^T bf16: [m][d]
  int tid = threadIdx.x, lane = tid & 31, w = tid >> 5;
  int bh = blockIdx.x / NC, n = blockIdx.x % NC;
  const size_t src = ((size_t)bh * Ll + (size_t)n * CH) * Dd;
  // Q/K: padded row copies (async to LDS when available)
  for (int i = tid; i < CH * 8; i += 256) {
    int l = i >> 3, seg = (i & 7) * 8;
#if HAVE_ASYNC
    __builtin_amdgcn_global_load_async_to_lds_b128(
        to_glb(qm + src + (size_t)l * Dd + seg), to_lds(Qs + l * QLD + seg), 0, 0);
    __builtin_amdgcn_global_load_async_to_lds_b128(
        to_glb(kb + src + (size_t)l * Dd + seg), to_lds(Ks + l * QLD + seg), 0, 0);
#else
    *(u32x4*)(Qs + l * QLD + seg) = *(const u32x4*)(qm + src + (size_t)l * Dd + seg);
    *(u32x4*)(Ks + l * QLD + seg) = *(const u32x4*)(kb + src + (size_t)l * Dd + seg);
#endif
  }
  stage_transposed(vm + src, Vt, tid);
  // exclusive prefix state (kvT layout [m][d]); accumulate f32, store bf16 once
  const float* kvbase = kvws + (size_t)bh * NC * Dd * Dd;
  for (int i = tid; i < Dd * Dd; i += 256) {
    float s = 0.f;
    for (int j = 0; j < n; ++j) s += kvbase[(size_t)j * Dd * Dd + i];
    statebf[(i >> 6) * QLD + (i & 63)] = (bf16_t)s;
  }
  wait_async_done();
  __syncthreads();
  // Q A-fragments: loaded once, used by phase 1 (scores) and phase 2 (inter)
  v16bf aq0 = frag_a_rows(Qs + (16 * w) * QLD + 0, QLD, lane);
  v16bf aq1 = frag_a_rows(Qs + (16 * w) * QLD + 32, QLD, lane);
  // Phase 1: scores rows 16w..16w+15 (QK^T), mask tril, bf16 -> LDS
  {
    int coll = lane & 15, rhalf = (lane >> 4) & 1;
#pragma unroll
    for (int jt = 0; jt < 8; ++jt) {
      v16bf b0 = frag_b_rows(Ks + (jt * 16) * QLD + 0, QLD, lane);
      v16bf b1 = frag_b_rows(Ks + (jt * 16) * QLD + 32, QLD, lane);
      v8f s = {};
      s = wmma_bf16(aq0, b0, s);
      s = wmma_bf16(aq1, b1, s);
      int j = jt * 16 + coll;
#pragma unroll
      for (int rr = 0; rr < 8; ++rr) {
        int i = 16 * w + rhalf * 8 + rr;
        Sc[(size_t)i * CHP + j] = (bf16_t)(j <= i ? s[rr] : 0.0f);
      }
    }
  }
  __syncthreads();
  // Phase 2: out rows 16w..16w+15 = Q@state + Sc@V  (shared accumulators)
  v8f c[4] = {};
#pragma unroll
  for (int mt = 0; mt < 4; ++mt) {
    // B[kk=d,nn=m] = state[d][m] = statebf[m][d]
    v16bf b0 = frag_b_rows(statebf + (mt * 16) * QLD + 0, QLD, lane);
    v16bf b1 = frag_b_rows(statebf + (mt * 16) * QLD + 32, QLD, lane);
    c[mt] = wmma_bf16(aq0, b0, c[mt]);
    c[mt] = wmma_bf16(aq1, b1, c[mt]);
  }
#pragma unroll
  for (int j0 = 0; j0 < CH; j0 += 32) {
    v16bf a = frag_a_rows(Sc + (16 * w) * CHP + j0, CHP, lane);
#pragma unroll
    for (int mt = 0; mt < 4; ++mt) {
      // B[kk=l',nn=m] = V[l'][m] = Vt[m][l']
      v16bf b = frag_b_rows(Vt + (mt * 16) * CHP + j0, CHP, lane);
      c[mt] = wmma_bf16(a, b, c[mt]);
    }
  }
  // Epilogue: scale by sink_allocation, scatter to [B,L,E] bf16
  int coll = lane & 15, rhalf = (lane >> 4) & 1;
  int bb = bh >> 3, hh = bh & 7;
  float sa[8];
#pragma unroll
  for (int rr = 0; rr < 8; ++rr)
    sa[rr] = sink[(size_t)bh * Ll + n * CH + 16 * w + rhalf * 8 + rr];
#pragma unroll
  for (int mt = 0; mt < 4; ++mt) {
#pragma unroll
    for (int rr = 0; rr < 8; ++rr) {
      int l = n * CH + 16 * w + rhalf * 8 + rr;
      int e = hh * Dd + mt * 16 + coll;
      attnb[((size_t)bb * Ll + l) * Ee + e] = (bf16_t)(c[mt][rr] * sa[rr]);
    }
  }
}

// ---------------- Launcher ----------------
extern "C" void kernel_launch(void* const* d_in, const int* in_sizes, int n_in,
                              void* d_out, int out_size, void* d_ws, size_t ws_size,
                              hipStream_t stream) {
  const float* x  = (const float*)d_in[0];
  const float* Wq = (const float*)d_in[1];
  const float* bq = (const float*)d_in[2];
  const float* Wk = (const float*)d_in[3];
  const float* bk = (const float*)d_in[4];
  const float* Wv = (const float*)d_in[5];
  const float* bv = (const float*)d_in[6];
  const float* Wo = (const float*)d_in[7];
  const float* bo = (const float*)d_in[8];
  float* out = (float*)d_out;

  char* ws = (char*)d_ws;
  size_t off = 0;
  auto alloc = [&](size_t bytes) -> char* {
    char* p = ws + off;
    off += (bytes + 255) & ~(size_t)255;
    return p;
  };
  bf16_t* xbf = (bf16_t*)alloc((size_t)Nrows * Ee * 2);
  bf16_t* Wqb = (bf16_t*)alloc((size_t)Ee * Ee * 2);
  bf16_t* Wkb = (bf16_t*)alloc((size_t)Ee * Ee * 2);
  bf16_t* Wvb = (bf16_t*)alloc((size_t)Ee * Ee * 2);
  bf16_t* Wob = (bf16_t*)alloc((size_t)Ee * Ee * 2);
  float* qf = (float*)alloc((size_t)Nrows * Ee * 4);
  float* kf = (float*)alloc((size_t)Nrows * Ee * 4);
  float* vf = (float*)alloc((size_t)Nrows * Ee * 4);
  bf16_t* qmb = (bf16_t*)alloc((size_t)Nrows * Ee * 2);
  bf16_t* kbb = (bf16_t*)alloc((size_t)Nrows * Ee * 2);
  bf16_t* vmb = (bf16_t*)alloc((size_t)Nrows * Ee * 2);
  float* sinkp = (float*)alloc((size_t)Bq * Hh * Ll * 4);
  // qf/kf are dead after flow_scan (stream ordering) -> reuse their space.
  float*  kvws  = (float*)qf;   // [B*H, NC, 64, 64] f32 (kv^T per chunk)
  bf16_t* attnb = (bf16_t*)kf;  // [B, L, E] bf16

  const int xN = Nrows * Ee, wN = Ee * Ee;
  f32_to_bf16_kernel<<<(xN + 255) / 256, 256, 0, stream>>>(x, xbf, xN);
  f32_to_bf16_kernel<<<(wN + 255) / 256, 256, 0, stream>>>(Wq, Wqb, wN);
  f32_to_bf16_kernel<<<(wN + 255) / 256, 256, 0, stream>>>(Wk, Wkb, wN);
  f32_to_bf16_kernel<<<(wN + 255) / 256, 256, 0, stream>>>(Wv, Wvb, wN);
  f32_to_bf16_kernel<<<(wN + 255) / 256, 256, 0, stream>>>(Wo, Wob, wN);

  dim3 g(Nrows / 128, Ee / 64);
  gemm_bf16_kernel<true, 1><<<g, 256, 0, stream>>>(xbf, Wqb, bq, qf);
  gemm_bf16_kernel<true, 1><<<g, 256, 0, stream>>>(xbf, Wkb, bk, kf);
  gemm_bf16_kernel<true, 0><<<g, 256, 0, stream>>>(xbf, Wvb, bv, vf);

  flow_scan_kernel<<<Bq * Hh, 32, 0, stream>>>(qf, kf, vf, qmb, kbb, vmb, sinkp);

  kv_chunk_kernel<<<Bq * Hh * NC, 256, 0, stream>>>(kbb, vmb, kvws);

  const int attn_lds =
      CH * QLD * 2 * 2 + Dd * CHP * 2 + CH * CHP * 2 + Dd * QLD * 2;  // 96 KB
  attn_chunk_kernel<<<Bq * Hh * NC, 256, attn_lds, stream>>>(qmb, kbb, vmb, kvws,
                                                             sinkp, attnb);

  gemm_bf16_kernel<false, 0><<<g, 256, 0, stream>>>(attnb, Wob, bo, out);
}
